// MHA_18038862643618
// MI455X (gfx1250) — matrix operand order repaired and checked by reading, not compiled, not run on hardware
//
#include <hip/hip_runtime.h>
#include <stdint.h>

// ---------------------------------------------------------------------------
// MHA (B=2, T=2048, D=2048, H=16, Hd=128) for gfx1250 (MI455X, wave32, WMMA)
//   k1: qkv = x @ w_qkv            (bf16 WMMA, bf16 out, double-buffered LDS)
//   k2: rope + split heads         (q/k/v bf16, [B,H,T,128])
//   k3: flash attention (causal)   (bf16 WMMA, online softmax, dbl-buffered)
//   k4: out = attn @ w_out         (bf16 WMMA, f32 out)
// Async global->LDS (CDNA5, ASYNCcnt) used where data is already bf16.
// ---------------------------------------------------------------------------

#define B_SZ 2
#define T_SZ 2048
#define D_SZ 2048
#define H_SZ 16
#define HD_SZ 128

typedef __attribute__((ext_vector_type(16))) __bf16 v16bf;
typedef __attribute__((ext_vector_type(8)))  float  v8f;
typedef __attribute__((ext_vector_type(4)))  int    v4i;

typedef v4i __attribute__((address_space(1)))* gv4ip;   // global v4i*
typedef v4i __attribute__((address_space(3)))* lv4ip;   // LDS v4i*

static __device__ __forceinline__ unsigned short f2bf(float f) {
  unsigned int u = __float_as_uint(f);
  u += 0x7FFFu + ((u >> 16) & 1u);            // round-to-nearest-even
  return (unsigned short)(u >> 16);
}
static __device__ __forceinline__ float bf2f(unsigned short h) {
  return __uint_as_float(((unsigned int)h) << 16);
}

union Frag16 {                  // one wave32 16-bit WMMA operand (16 bf16/lane)
  v16bf v;
  struct { uint4 lo, hi; } q;
};

// ---- CDNA5 async global->LDS (ASYNCcnt path), guarded fallback ------------
#if __has_builtin(__builtin_amdgcn_global_load_async_to_lds_b128)
#define HAVE_ASYNC 1
#else
#define HAVE_ASYNC 0
#endif

static __device__ __forceinline__ void cp_b128(unsigned short* ldst,
                                               const unsigned short* gsrc) {
#if HAVE_ASYNC
  __builtin_amdgcn_global_load_async_to_lds_b128(
      (gv4ip)(uintptr_t)gsrc,
      (lv4ip)(unsigned)(uintptr_t)ldst,
      0, 0);
#else
  *(uint4*)ldst = *(const uint4*)gsrc;
#endif
}
static __device__ __forceinline__ void wait_async() {
#if HAVE_ASYNC
  asm volatile("s_wait_asynccnt 0" ::: "memory");
#endif
}

// ---------------------------------------------------------------------------
// Tiled WMMA GEMM: C[M,N] = A[M,K] * B[K,N].
// Block tile 128x128, BK=32, 256 threads = 8 waves, wave tile 32x64
// (2x4 WMMA 16x16x32 bf16, f32 accum). Double-buffered LDS, 1 barrier/step.
// A fp32 (convert) or bf16 (async copy); B fp32 staged transposed as bf16.
// ---------------------------------------------------------------------------
template<bool A_BF16, bool OUT_BF16>
__global__ __launch_bounds__(256)
void gemm_wmma(const void* __restrict__ Ap, const float* __restrict__ Bp,
               void* __restrict__ Cp, int M, int N, int K)
{
  __shared__ alignas(16) unsigned short sA [2][128][40];   // [buf][m][k]
  __shared__ alignas(16) unsigned short sBT[2][128][40];   // [buf][n][k]

  const int tid   = threadIdx.x;
  const int lane  = tid & 31;
  const int wid   = tid >> 5;
  const int waveM = wid & 3;            // 4 row-groups of 32
  const int waveN = wid >> 2;           // 2 col-groups of 64
  const int m0    = blockIdx.y * 128;
  const int n0    = blockIdx.x * 128;
  const int l16   = lane & 15;
  const int lhalf = lane >> 4;

  const v8f vzero = {};
  v8f acc[2][4];
#pragma unroll
  for (int i = 0; i < 2; ++i)
#pragma unroll
    for (int j = 0; j < 4; ++j) acc[i][j] = vzero;

  auto stage = [&](int buf, int k0) {
    // ---- A tile (128 x 32) ---------------------------------------------
    if (A_BF16) {
      const unsigned short* A16 = (const unsigned short*)Ap;
#pragma unroll
      for (int i = 0; i < 2; ++i) {               // 512 x b128
        int v   = tid + i * 256;
        int row = v >> 2;
        int c8  = (v & 3) << 3;
        cp_b128(&sA[buf][row][c8], A16 + (size_t)(m0 + row) * K + k0 + c8);
      }
    } else {
      const float* Af = (const float*)Ap;
#pragma unroll
      for (int i = 0; i < 4; ++i) {               // 1024 x vec4, cvt to bf16
        int v   = tid + i * 256;
        int row = v >> 3;
        int c4  = (v & 7) << 2;
        float4 d = *(const float4*)(Af + (size_t)(m0 + row) * K + k0 + c4);
        uint2 pk;
        pk.x = (unsigned)f2bf(d.x) | ((unsigned)f2bf(d.y) << 16);
        pk.y = (unsigned)f2bf(d.z) | ((unsigned)f2bf(d.w) << 16);
        *(uint2*)&sA[buf][row][c4] = pk;
      }
    }
    // ---- B tile (32 x 128), transposed, cvt to bf16 --------------------
#pragma unroll
    for (int i = 0; i < 4; ++i) {                 // 1024 x vec4
      int v  = tid + i * 256;
      int kr = v >> 5;
      int c4 = (v & 31) << 2;
      float4 d = *(const float4*)(Bp + (size_t)(k0 + kr) * N + n0 + c4);
      sBT[buf][c4 + 0][kr] = f2bf(d.x);
      sBT[buf][c4 + 1][kr] = f2bf(d.y);
      sBT[buf][c4 + 2][kr] = f2bf(d.z);
      sBT[buf][c4 + 3][kr] = f2bf(d.w);
    }
  };

  auto compute = [&](int buf) {
    Frag16 a[2], b[4];
    const int ka  = lhalf * 8;    // A frag: lanes0-15 K{0..7,16..23}; +8 hi
    const int kbo = lhalf * 8;    // B frag: lanes0-15 K{0..7,16..23}; +8 hi
#pragma unroll
    for (int mi = 0; mi < 2; ++mi) {
      int row = waveM * 32 + mi * 16 + l16;
      a[mi].q.lo = *(const uint4*)&sA[buf][row][ka];
      a[mi].q.hi = *(const uint4*)&sA[buf][row][ka + 16];
    }
#pragma unroll
    for (int ni = 0; ni < 4; ++ni) {
      int col = waveN * 64 + ni * 16 + l16;
      b[ni].q.lo = *(const uint4*)&sBT[buf][col][kbo];
      b[ni].q.hi = *(const uint4*)&sBT[buf][col][kbo + 16];
    }
#pragma unroll
    for (int mi = 0; mi < 2; ++mi)
#pragma unroll
      for (int ni = 0; ni < 4; ++ni)
        acc[mi][ni] = __builtin_amdgcn_wmma_f32_16x16x32_bf16(
            false, a[mi].v, false, b[ni].v, (short)0, acc[mi][ni], false, false);
  };

  stage(0, 0);
  wait_async();
  __syncthreads();
  for (int k0 = 0; k0 < K; k0 += 32) {
    int cur = (k0 >> 5) & 1;
    if (k0 + 32 < K) stage(cur ^ 1, k0 + 32);   // overlap next tile with math
    compute(cur);
    wait_async();
    __syncthreads();
  }

  // ---- epilogue: lane holds (M = r + 8*lhalf, N = l16) ------------------
#pragma unroll
  for (int mi = 0; mi < 2; ++mi) {
#pragma unroll
    for (int ni = 0; ni < 4; ++ni) {
      int gcol = n0 + waveN * 64 + ni * 16 + l16;
#pragma unroll
      for (int r = 0; r < 8; ++r) {
        int grow = m0 + waveM * 32 + mi * 16 + lhalf * 8 + r;
        float val = acc[mi][ni][r];
        if (OUT_BF16)
          ((unsigned short*)Cp)[(size_t)grow * N + gcol] = f2bf(val);
        else
          ((float*)Cp)[(size_t)grow * N + gcol] = val;
      }
    }
  }
}

// ---------------------------------------------------------------------------
// RoPE + split heads: qkv bf16 [B,T,3D] -> q/k/v bf16 [B,H,T,128]
// ---------------------------------------------------------------------------
__global__ __launch_bounds__(128)
void rope_split(const unsigned short* __restrict__ qkv,
                unsigned short* __restrict__ qb,
                unsigned short* __restrict__ kb,
                unsigned short* __restrict__ vb)
{
  const int t   = blockIdx.x;
  const int b   = blockIdx.y;
  const int tid = threadIdx.x;
  const unsigned short* row = qkv + ((size_t)(b * T_SZ + t)) * (3 * D_SZ);
  const float ft = (float)t;
  const float NLOGB = -9.210340371976184f / 64.0f;   // -ln(10000)/half

#pragma unroll
  for (int i = 0; i < 8; ++i) {              // 16 heads * 64 rotation pairs
    int idx = tid + i * 128;
    int h = idx >> 6;
    int p = idx & 63;
    int e0 = 2 * p, e1 = 2 * p + 1;
    float if0 = __expf(NLOGB * (float)(e0 & 63));
    float if1 = __expf(NLOGB * (float)(e1 & 63));
    float a0 = ft * if0, a1 = ft * if1;
    float c0 = __cosf(a0), sn0 = __sinf(a0);
    float c1 = __cosf(a1), sn1 = __sinf(a1);
    size_t dst = ((size_t)(b * H_SZ + h) * T_SZ + t) * HD_SZ;
    {   // Q
      float xe = bf2f(row[h * HD_SZ + e0]);
      float xo = bf2f(row[h * HD_SZ + e1]);
      qb[dst + e0] = f2bf(xe * c0 - xo * sn0);
      qb[dst + e1] = f2bf(xe * sn1 + xo * c1);
    }
    {   // K
      float xe = bf2f(row[D_SZ + h * HD_SZ + e0]);
      float xo = bf2f(row[D_SZ + h * HD_SZ + e1]);
      kb[dst + e0] = f2bf(xe * c0 - xo * sn0);
      kb[dst + e1] = f2bf(xe * sn1 + xo * c1);
    }
  }
#pragma unroll
  for (int i = 0; i < 16; ++i) {             // V: plain relayout
    int idx = tid + i * 128;
    int h = idx >> 7;
    int d = idx & 127;
    vb[((size_t)(b * H_SZ + h) * T_SZ + t) * HD_SZ + d] = row[2 * D_SZ + idx];
  }
}

// ---------------------------------------------------------------------------
// Flash attention (causal). grid = (T/64, B*H), 128 threads = 4 waves.
// Wave w owns 16 q rows; 32-key blocks, double-buffered K/V^T, online softmax.
// ---------------------------------------------------------------------------
__global__ __launch_bounds__(128)
void flash_attn(const unsigned short* __restrict__ qb,
                const unsigned short* __restrict__ kb,
                const unsigned short* __restrict__ vb,
                unsigned short* __restrict__ ao)
{
  __shared__ alignas(16) unsigned short sQ [64][136];      // [q][hd]
  __shared__ alignas(16) unsigned short sK [2][32][136];   // [buf][key][hd]
  __shared__ alignas(16) unsigned short sVT[2][128][40];   // [buf][hd][key]
  __shared__ alignas(16) unsigned short sP [4][16][40];    // wave-private P

  const int tid   = threadIdx.x;
  const int lane  = tid & 31;
  const int wid   = tid >> 5;
  const int l16   = lane & 15;
  const int lhalf = lane >> 4;
  const int qtile = blockIdx.x;
  const int bh    = blockIdx.y;
  const int b     = bh >> 4;
  const int h     = bh & 15;
  const float scale = 0.08838834764831845f;   // 1/sqrt(128)
  const size_t headBase = (size_t)bh * T_SZ * HD_SZ;

  auto stageKV = [&](int buf, int j) {
#pragma unroll
    for (int i = 0; i < 4; ++i) {             // 512 x b128
      int v  = tid + i * 128;
      int kr = v >> 4;
      int d8 = (v & 15) << 3;
      cp_b128(&sK[buf][kr][d8],
              kb + headBase + (size_t)(j * 32 + kr) * HD_SZ + d8);
      union { uint4 q; unsigned short u[8]; } dv;   // V transposed via VGPRs
      dv.q = *(const uint4*)(vb + headBase + (size_t)(j * 32 + kr) * HD_SZ + d8);
#pragma unroll
      for (int e = 0; e < 8; ++e) sVT[buf][d8 + e][kr] = dv.u[e];
    }
  };

  // stage Q tile (64 x 128) once (async path when available)
#pragma unroll
  for (int i = 0; i < 8; ++i) {               // 1024 x b128
    int v  = tid + i * 128;
    int qr = v >> 4;
    int d8 = (v & 15) << 3;
    cp_b128(&sQ[qr][d8],
            qb + headBase + (size_t)(qtile * 64 + qr) * HD_SZ + d8);
  }
  stageKV(0, 0);
  wait_async();
  __syncthreads();

  const v8f vzero = {};
  float mst[8], lst[8];
  v8f o[8];
#pragma unroll
  for (int r = 0; r < 8; ++r) { mst[r] = -1e30f; lst[r] = 0.f; }
#pragma unroll
  for (int c = 0; c < 8; ++c) o[c] = vzero;

  const int qbase   = qtile * 64 + wid * 16;
  const int nblocks = (qtile + 1) * 2;        // causal block bound

  for (int j = 0; j < nblocks; ++j) {
    const int cur = j & 1;
    if (j + 1 < nblocks) stageKV(cur ^ 1, j + 1);  // overlap next K/V block

    // ---- S = Q K^T : 16 q rows x 32 keys (2 tiles x 4 hd-chunks) -------
    float fs[2][8];
#pragma unroll
    for (int nt = 0; nt < 2; ++nt) {
      v8f s = vzero;
#pragma unroll
      for (int hc = 0; hc < 4; ++hc) {
        Frag16 aq, bk;
        int qr = wid * 16 + l16;
        int ka = hc * 32 + lhalf * 8;
        aq.q.lo = *(const uint4*)&sQ[qr][ka];
        aq.q.hi = *(const uint4*)&sQ[qr][ka + 16];
        int key = nt * 16 + l16;
        int kv  = hc * 32 + lhalf * 8;
        bk.q.lo = *(const uint4*)&sK[cur][key][kv];
        bk.q.hi = *(const uint4*)&sK[cur][key][kv + 16];
        s = __builtin_amdgcn_wmma_f32_16x16x32_bf16(
            false, aq.v, false, bk.v, (short)0, s, false, false);
      }
      int keycol = j * 32 + nt * 16 + l16;
#pragma unroll
      for (int r = 0; r < 8; ++r) {
        int qrow = qbase + lhalf * 8 + r;
        float sv = s[r] * scale;
        fs[nt][r] = (keycol > qrow) ? -1e30f : sv;   // causal mask
      }
    }

    // ---- online softmax (row lives across a 16-lane half-wave) ---------
#pragma unroll
    for (int r = 0; r < 8; ++r) {
      float mb = fmaxf(fs[0][r], fs[1][r]);
#pragma unroll
      for (int off = 1; off < 16; off <<= 1)
        mb = fmaxf(mb, __shfl_xor(mb, off, 32));
      float mnew  = fmaxf(mst[r], mb);
      float alpha = __expf(mst[r] - mnew);
      float p0 = __expf(fs[0][r] - mnew);
      float p1 = __expf(fs[1][r] - mnew);
      float rs = p0 + p1;
#pragma unroll
      for (int off = 1; off < 16; off <<= 1)
        rs += __shfl_xor(rs, off, 32);
      lst[r] = lst[r] * alpha + rs;
      mst[r] = mnew;
#pragma unroll
      for (int c = 0; c < 8; ++c) o[c][r] *= alpha;
      int mrow = lhalf * 8 + r;                    // re-fragment P via LDS
      sP[wid][mrow][l16]      = f2bf(p0);
      sP[wid][mrow][16 + l16] = f2bf(p1);
    }
    asm volatile("s_wait_dscnt 0" ::: "memory");   // P stores -> frag loads

    // ---- O += P @ V : 8 WMMAs (K = 32 keys) ----------------------------
#pragma unroll
    for (int c = 0; c < 8; ++c) {
      Frag16 pa, pv;
      int ka = lhalf * 8;
      pa.q.lo = *(const uint4*)&sP[wid][l16][ka];
      pa.q.hi = *(const uint4*)&sP[wid][l16][ka + 16];
      int n  = c * 16 + l16;
      int kv = lhalf * 8;
      pv.q.lo = *(const uint4*)&sVT[cur][n][kv];
      pv.q.hi = *(const uint4*)&sVT[cur][n][kv + 16];
      o[c] = __builtin_amdgcn_wmma_f32_16x16x32_bf16(
          false, pa.v, false, pv.v, (short)0, o[c], false, false);
    }
    wait_async();
    __syncthreads();
  }

  // ---- normalize and write attn output bf16 as [B,T,D] -----------------
#pragma unroll
  for (int r = 0; r < 8; ++r) {
    float inv = 1.0f / lst[r];
    int t = qbase + lhalf * 8 + r;
#pragma unroll
    for (int c = 0; c < 8; ++c) {
      int col = h * HD_SZ + c * 16 + l16;
      ao[((size_t)(b * T_SZ + t)) * D_SZ + col] = f2bf(o[c][r] * inv);
    }
  }
}

// ---------------------------------------------------------------------------
extern "C" void kernel_launch(void* const* d_in, const int* in_sizes, int n_in,
                              void* d_out, int out_size, void* d_ws, size_t ws_size,
                              hipStream_t stream)
{
  (void)in_sizes; (void)n_in; (void)out_size; (void)ws_size;
  const float* x     = (const float*)d_in[0];
  const float* w_qkv = (const float*)d_in[1];
  const float* w_out = (const float*)d_in[2];
  float* out = (float*)d_out;

  char* ws = (char*)d_ws;
  unsigned short* qkv = (unsigned short*)ws;                          // 48 MB
  unsigned short* qb  = (unsigned short*)(ws + (size_t)4096 * 6144 * 2);
  unsigned short* kb  = qb + (size_t)4096 * 2048;
  unsigned short* vb  = kb + (size_t)4096 * 2048;
  unsigned short* ao  = (unsigned short*)ws;    // reuse dead qkv region

  // 1) qkv = x @ w_qkv   (M=4096, N=6144, K=2048), bf16 out
  gemm_wmma<false, true><<<dim3(6144 / 128, 4096 / 128), dim3(256), 0, stream>>>(
      x, w_qkv, qkv, 4096, 6144, 2048);
  // 2) RoPE + head split
  rope_split<<<dim3(T_SZ, B_SZ), dim3(128), 0, stream>>>(qkv, qb, kb, vb);
  // 3) flash attention
  flash_attn<<<dim3(T_SZ / 64, B_SZ * H_SZ), dim3(128), 0, stream>>>(qb, kb, vb, ao);
  // 4) out = attn @ w_out (M=4096, N=2048, K=2048), f32 out
  gemm_wmma<true, false><<<dim3(2048 / 128, 4096 / 128), dim3(256), 0, stream>>>(
      ao, w_out, out, 4096, 2048, 2048);
}
